// BiLSTM_53300544143472
// MI455X (gfx1250) — matrix operand order repaired and checked by reading, not compiled, not run on hardware
//
#include <hip/hip_runtime.h>
#include <hip/hip_fp16.h>

// ---------------------------------------------------------------------------
// BiLSTM LM: embed -> 2x fwd LSTM, 2x bwd LSTM -> concat -> vocab projection
// V=32000 E=512 H=512 L=2 B=16 T=256
// Strategy: f16 WMMA (v_wmma_f32_16x16x32_f16) for all GEMMs; persistent
// single-workgroup recurrence with h in LDS and c in registers.
// ---------------------------------------------------------------------------

typedef __attribute__((ext_vector_type(16))) _Float16 v16h;
typedef __attribute__((ext_vector_type(8)))  _Float16 v8h;
typedef __attribute__((ext_vector_type(8)))  float    v8f;

#define WMMA_F16(a, b, c) \
    __builtin_amdgcn_wmma_f32_16x16x32_f16(false, (a), false, (b), (short)0, (c), false, false)

// A operand (16x32, f16, M x K), source row-major with leading dim lda.
// lane<16 holds row m=lane, K=k0..k0+15;
// lane>=16 holds row m=lane-16, K=k0+16..k0+31 (matches B operand order).
template <typename PTR>
__device__ __forceinline__ v16h load_a_16x32(PTR src, int lda, int m0, int k0, int lane) {
    int row = m0 + (lane & 15);
    int kb  = k0 + ((lane >> 4) << 4);          // +16 for upper half-wave
    union { v16h v; v8h h[2]; } u;
    u.h[0] = *(const v8h*)(src + (size_t)row * lda + kb);
    u.h[1] = *(const v8h*)(src + (size_t)row * lda + kb + 8);
    return u.v;
}

// B operand (32x16, f16, K x N) where B = W^T and W is row-major (N x K), ld ldw.
// lane<16 holds column n=n0+lane, K=k0..k0+15 (contiguous in W row);
// lane>=16 holds column n=n0+lane-16, K=k0+16..k0+31.
__device__ __forceinline__ v16h load_b_w_t(const _Float16* __restrict__ W, int ldw,
                                           int n0, int k0, int lane) {
    int col = n0 + (lane & 15);
    int kb  = k0 + ((lane >> 4) << 4);          // +16 for upper half-wave
    return *(const v16h*)(W + (size_t)col * ldw + kb);
}

__device__ __forceinline__ float sigf(float x) { return 1.0f / (1.0f + __expf(-x)); }

// ---------------------------------------------------------------------------
// Elementwise f32 -> f16 weight conversion
// ---------------------------------------------------------------------------
__global__ void cvt_f32_f16(const float* __restrict__ in, _Float16* __restrict__ out, int n) {
    int i = blockIdx.x * blockDim.x + threadIdx.x;
    if (i < n) out[i] = (_Float16)in[i];
}

// ---------------------------------------------------------------------------
// Embedding gather: xs[(t*B + b)*E + e] = emb[x[b*T + t]*E + e]  (f16)
// ---------------------------------------------------------------------------
__global__ void embed_gather(const int* __restrict__ x, const float* __restrict__ emb,
                             _Float16* __restrict__ xs, int Bn, int Tn, int En) {
    int i = blockIdx.x * blockDim.x + threadIdx.x;
    int total = Bn * Tn * En;
    if (i >= total) return;
    int e = i % En;
    int r = i / En;          // r = t*Bn + b
    int b = r % Bn;
    int t = r / Bn;
    int tok = x[b * Tn + t];
    xs[i] = (_Float16)emb[(size_t)tok * En + e];
}

// ---------------------------------------------------------------------------
// WMMA GEMM:  C(M,N) = A(M,K) @ W(N,K)^T + bias1 + bias2      (f16 in, f32 out)
// 256 threads = 8 waves; wave tile 32x32 (2x2 WMMA); WG tile 64x128.
// grid = (N/128, M/64). swapTB remaps output row r=t*Bn+b -> b*Tn+t (logits).
// ---------------------------------------------------------------------------
__global__ __launch_bounds__(256) void gemm_wmma_f16(
    const _Float16* __restrict__ A, int lda,
    const _Float16* __restrict__ W, int ldw,
    const float* __restrict__ bias1, const float* __restrict__ bias2,
    float* __restrict__ C, int ldc,
    int Kdim, int swapTB, int Bn, int Tn) {
    int lane = threadIdx.x & 31;
    int wave = threadIdx.x >> 5;
    int wr = wave & 1, wc = wave >> 1;
    int m0 = blockIdx.y * 64 + wr * 32;
    int n0 = blockIdx.x * 128 + wc * 32;

    v8f acc00 = {}, acc01 = {}, acc10 = {}, acc11 = {};
    for (int k0 = 0; k0 < Kdim; k0 += 32) {
        v16h a0 = load_a_16x32(A, lda, m0,      k0, lane);
        v16h a1 = load_a_16x32(A, lda, m0 + 16, k0, lane);
        v16h b0 = load_b_w_t(W, ldw, n0,      k0, lane);
        v16h b1 = load_b_w_t(W, ldw, n0 + 16, k0, lane);
        acc00 = WMMA_F16(a0, b0, acc00);
        acc01 = WMMA_F16(a0, b1, acc01);
        acc10 = WMMA_F16(a1, b0, acc10);
        acc11 = WMMA_F16(a1, b1, acc11);
    }

    // C/D layout: lane holds column n=lane%16; rows m = 8*(lane/16) + v.
    int nloc = lane & 15;
    int mb   = (lane >> 4) << 3;
    int cols[2] = { n0 + nloc, n0 + 16 + nloc };
    float badd[2];
#pragma unroll
    for (int ni = 0; ni < 2; ++ni) {
        float bb = bias1 ? bias1[cols[ni]] : 0.0f;
        if (bias2) bb += bias2[cols[ni]];
        badd[ni] = bb;
    }
    const v8f* accs[2][2] = { { &acc00, &acc01 }, { &acc10, &acc11 } };
#pragma unroll
    for (int mi = 0; mi < 2; ++mi) {
#pragma unroll
        for (int v = 0; v < 8; ++v) {
            int r = m0 + mi * 16 + mb + v;
            int orow = swapTB ? ((r % Bn) * Tn + r / Bn) : r;
#pragma unroll
            for (int ni = 0; ni < 2; ++ni)
                C[(size_t)orow * ldc + cols[ni]] = (*accs[mi][ni])[v] + badd[ni];
        }
    }
}

// ---------------------------------------------------------------------------
// Persistent LSTM recurrence (one workgroup, 32 waves, H=512, B=16, 4H=2048).
// Wave w owns hidden slice j in [w*16, w*16+16) for ALL four gates, so the
// LSTM cell update runs entirely in registers (c state never leaves VGPRs).
// g = xproj[t] + h @ Whh^T  (biases already folded into xproj).
// ---------------------------------------------------------------------------
__global__ __launch_bounds__(1024) void lstm_recurrence(
    const _Float16* __restrict__ Whh,   // (2048, 512) f16 row-major
    const float*    __restrict__ xproj, // (T*B, 2048) f32, includes bih+bhh
    _Float16*       __restrict__ hout,  // (T*B, ldo) f16, write cols [ocol, ocol+512)
    int ldo, int ocol, int Tdim, int reverse) {
    const int Hh = 512, Bn = 16, G = 2048;
    __shared__ _Float16 h_lds[16 * 512];

    int tid  = threadIdx.x;
    int lane = tid & 31;
    int wave = tid >> 5;
    for (int i = tid; i < Bn * Hh; i += 1024) h_lds[i] = (_Float16)0.0f;
    __syncthreads();

    int jloc = lane & 15;
    int j    = wave * 16 + jloc;        // hidden index owned by this lane
    int mb   = (lane >> 4) << 3;        // batch-row base for this half-wave

    v8f c_reg = {};                     // cell state: 8 batches x (this j)
    for (int s = 0; s < Tdim; ++s) {
        int t = reverse ? (Tdim - 1 - s) : s;
        const float* xp = xproj + (size_t)t * Bn * G;

        v8f acc[4] = { v8f{}, v8f{}, v8f{}, v8f{} };   // i, f, g, o gate tiles
        for (int k0 = 0; k0 < Hh; k0 += 32) {
            v16h a = load_a_16x32(h_lds, Hh, 0, k0, lane);
#pragma unroll
            for (int g = 0; g < 4; ++g) {
                v16h b = load_b_w_t(Whh, Hh, g * 512 + wave * 16, k0, lane);
                acc[g] = WMMA_F16(a, b, acc[g]);
            }
        }
        __syncthreads();                // all reads of h_lds for this step done

#pragma unroll
        for (int v = 0; v < 8; ++v) {
            int m = mb + v;             // batch index
            const float* xr = xp + (size_t)m * G;
            float ig = acc[0][v] + xr[j];
            float fg = acc[1][v] + xr[512 + j];
            float gg = acc[2][v] + xr[1024 + j];
            float og = acc[3][v] + xr[1536 + j];
            float cn = sigf(fg) * c_reg[v] + sigf(ig) * tanhf(gg);
            float hn = sigf(og) * tanhf(cn);
            c_reg[v] = cn;
            _Float16 h16 = (_Float16)hn;
            h_lds[m * Hh + j] = h16;
            hout[(size_t)(t * Bn + m) * ldo + ocol + j] = h16;
        }
        __syncthreads();                // h_lds updated before next step
    }
}

// ---------------------------------------------------------------------------
// Host-side orchestration
// ---------------------------------------------------------------------------
extern "C" void kernel_launch(void* const* d_in, const int* in_sizes, int n_in,
                              void* d_out, int out_size, void* d_ws, size_t ws_size,
                              hipStream_t stream) {
    (void)in_sizes; (void)n_in; (void)out_size; (void)ws_size;
    const int V = 32000, E = 512, H = 512, Bn = 16, Tn = 256;
    const int M = Bn * Tn;              // 4096 rows (t-major: r = t*B + b)
    const int G = 4 * H;                // 2048 gate width

    const int*   x        = (const int*)d_in[0];
    const float* emb      = (const float*)d_in[1];
    const float* Wih[4]   = { (const float*)d_in[2],  (const float*)d_in[6],
                              (const float*)d_in[10], (const float*)d_in[14] };
    const float* Whh[4]   = { (const float*)d_in[3],  (const float*)d_in[7],
                              (const float*)d_in[11], (const float*)d_in[15] };
    const float* bih[4]   = { (const float*)d_in[4],  (const float*)d_in[8],
                              (const float*)d_in[12], (const float*)d_in[16] };
    const float* bhh[4]   = { (const float*)d_in[5],  (const float*)d_in[9],
                              (const float*)d_in[13], (const float*)d_in[17] };
    const float* outW     = (const float*)d_in[18];
    const float* outb     = (const float*)d_in[19];
    float* logits         = (float*)d_out;

    // Workspace carve-up (256B aligned)
    char* ws = (char*)d_ws;
    size_t off = 0;
    auto carve = [&](size_t bytes) -> char* {
        char* p = ws + off;
        off = (off + bytes + 255) & ~(size_t)255;
        return p;
    };
    _Float16* xs16    = (_Float16*)carve((size_t)M * E * 2);        //  4 MB
    _Float16* wih16[4], *whh16[4];
    for (int i = 0; i < 4; ++i) wih16[i] = (_Float16*)carve((size_t)G * H * 2);
    for (int i = 0; i < 4; ++i) whh16[i] = (_Float16*)carve((size_t)G * H * 2);
    _Float16* outw16  = (_Float16*)carve((size_t)V * 2 * H * 2);    // 62.5 MB
    float*    xprojb  = (float*)   carve((size_t)M * G * 4);        // 32 MB
    _Float16* h0buf   = (_Float16*)carve((size_t)M * H * 2);        //  4 MB
    _Float16* hcat    = (_Float16*)carve((size_t)M * 2 * H * 2);    //  8 MB

    // 1. Weight conversions to f16
    {
        int n = G * H;
        dim3 blk(256), grd((n + 255) / 256);
        for (int i = 0; i < 4; ++i) {
            cvt_f32_f16<<<grd, blk, 0, stream>>>(Wih[i], wih16[i], n);
            cvt_f32_f16<<<grd, blk, 0, stream>>>(Whh[i], whh16[i], n);
        }
        int nw = V * 2 * H;
        cvt_f32_f16<<<dim3((nw + 255) / 256), blk, 0, stream>>>(outW, outw16, nw);
    }

    // 2. Embedding gather -> xs16 (t-major, f16)
    {
        int n = M * E;
        embed_gather<<<dim3((n + 255) / 256), dim3(256), 0, stream>>>(x, emb, xs16, Bn, Tn, E);
    }

    // 3. Four layer passes: xproj GEMM then persistent recurrence.
    //    chain layout: {fwd0, fwd1, bwd0, bwd1} = weight indices {0,1,2,3}
    struct Pass { int wi; const _Float16* ain; int lda; _Float16* hdst; int ldo, ocol, rev; };
    Pass passes[4] = {
        { 0, xs16,  E, h0buf, H,     0,   0 },   // fwd0
        { 1, h0buf, H, hcat,  2 * H, 0,   0 },   // fwd1 -> concat cols [0,512)
        { 2, xs16,  E, h0buf, H,     0,   1 },   // bwd0 (time reversed)
        { 3, h0buf, H, hcat,  2 * H, 512, 1 },   // bwd1 -> concat cols [512,1024)
    };
    for (int p = 0; p < 4; ++p) {
        const Pass& P = passes[p];
        // x_proj = A @ Wih^T + (bih + bhh)
        gemm_wmma_f16<<<dim3(G / 128, M / 64), dim3(256), 0, stream>>>(
            P.ain, P.lda, wih16[P.wi], H, bih[P.wi], bhh[P.wi],
            xprojb, G, H, /*swapTB=*/0, Bn, Tn);
        // serial recurrence over T steps in one workgroup
        lstm_recurrence<<<dim3(1), dim3(1024), 0, stream>>>(
            whh16[P.wi], xprojb, P.hdst, P.ldo, P.ocol, Tn, P.rev);
    }

    // 4. logits(b,t,:) = hcat(t,b,:) @ outW^T + out_b  (row swap fused in store)
    gemm_wmma_f16<<<dim3(V / 128, M / 64), dim3(256), 0, stream>>>(
        hcat, 2 * H, outw16, 2 * H, outb, (const float*)nullptr,
        logits, V, 2 * H, /*swapTB=*/1, Bn, Tn);
}
